// MoE_Layer_21457656611083
// MI455X (gfx1250) — matrix operand order repaired and checked by reading, not compiled, not run on hardware
//
#include <hip/hip_runtime.h>
#include <math.h>

typedef __bf16 v16bf __attribute__((ext_vector_type(16)));
typedef float  v8f   __attribute__((ext_vector_type(8)));
typedef unsigned int u32x4 __attribute__((ext_vector_type(4)));

#define NTOK   8192
#define DDIM   1024
#define NEXP   16
#define BM     128
#define BN     128
#define BK     32
#define KSTEPS (DDIM / BK)          // 32
#define LDSS   40                   // bf16 row stride: 32 + 8 pad -> 20 dwords
#define LDSSW  20                   // dword row stride

// pack two fp32 -> {bf16(hi)<<16 | bf16(lo)}; round-half-up via +0x8000,
// then one v_perm_b32 splices the two high halves (3 VALU total).
__device__ inline unsigned pack2bf(float lo, float hi) {
    unsigned a = __float_as_uint(lo) + 0x8000u;
    unsigned b = __float_as_uint(hi) + 0x8000u;
    // src0=b supplies bytes 4-7 of {b,a}; dst = {b[3],b[2],a[3],a[2]}
    return __builtin_amdgcn_perm(b, a, 0x07060302u);
}
__device__ inline v16bf load_bf16x16(const unsigned short* p, int off_elems) {
    struct V16S { u32x4 lo, hi; } s;
    s.lo = *(const u32x4*)(p);
    s.hi = *(const u32x4*)(p + off_elems);
    return __builtin_bit_cast(v16bf, s);
}

// ---------------------------------------------------------------- gating ----
__global__ __launch_bounds__(256)
void gate_topk_kernel(const float* __restrict__ x, const float* __restrict__ Wg,
                      const float* __restrict__ bg, int* __restrict__ counts,
                      float* __restrict__ imp, int* __restrict__ sel_e,
                      float* __restrict__ sel_w)
{
    __shared__ float lg[16][17];
    const int t  = threadIdx.x;
    const int tl = t >> 4;
    const int e  = t & 15;
    const int token = blockIdx.x * 16 + tl;

    float acc = bg[e];
    const float* xr = x + (size_t)token * DDIM;
    for (int i = 0; i < DDIM; i += 4) {
        float4 xv = *(const float4*)(xr + i);
        acc = fmaf(xv.x, Wg[(i + 0) * NEXP + e], acc);
        acc = fmaf(xv.y, Wg[(i + 1) * NEXP + e], acc);
        acc = fmaf(xv.z, Wg[(i + 2) * NEXP + e], acc);
        acc = fmaf(xv.w, Wg[(i + 3) * NEXP + e], acc);
    }
    lg[tl][e] = acc;
    __syncthreads();

    if (t < 16) {
        const int tok = blockIdx.x * 16 + t;
        float mx = -1e30f;
        #pragma unroll
        for (int k = 0; k < NEXP; ++k) mx = fmaxf(mx, lg[t][k]);
        float p[NEXP]; float s = 0.f;
        #pragma unroll
        for (int k = 0; k < NEXP; ++k) { p[k] = expf(lg[t][k] - mx); s += p[k]; }
        const float inv = 1.f / s;
        float v0 = -1e30f, v1 = -1e30f; int i0 = 0, i1 = 1;
        #pragma unroll
        for (int k = 0; k < NEXP; ++k) {
            float v = lg[t][k];
            if (v > v0)      { v1 = v0; i1 = i0; v0 = v; i0 = k; }
            else if (v > v1) { v1 = v;  i1 = k; }
        }
        #pragma unroll
        for (int k = 0; k < NEXP; ++k) lg[t][k] = p[k] * inv;
        const float w0 = 1.f / (1.f + expf(v1 - v0));
        const float w1 = 1.f - w0;
        sel_e[2 * tok]     = i0; sel_w[2 * tok]     = w0;
        sel_e[2 * tok + 1] = i1; sel_w[2 * tok + 1] = w1;
        atomicAdd(&counts[i0], 1);
        atomicAdd(&counts[i1], 1);
    }
    __syncthreads();
    if (t < 16) {
        float s = 0.f;
        #pragma unroll
        for (int k = 0; k < 16; ++k) s += lg[k][t];
        atomicAdd(&imp[t], s);
    }
}

// ------------------------------------------------------------------ scan ----
__global__ void scan_kernel(const int* __restrict__ counts, int* __restrict__ offsets)
{
    if (threadIdx.x == 0) {
        int acc = 0;
        for (int e = 0; e < NEXP; ++e) { offsets[e] = acc; acc += counts[e]; }
        offsets[NEXP] = acc;
    }
}

// --------------------------------------------------------------- scatter ----
__global__ __launch_bounds__(256)
void scatter_kernel(const int* __restrict__ sel_e, const float* __restrict__ sel_w,
                    const int* __restrict__ offsets, int* __restrict__ cursors,
                    int* __restrict__ perm_tok, float* __restrict__ perm_w)
{
    int n = blockIdx.x * blockDim.x + threadIdx.x;
    if (n >= NTOK) return;
    #pragma unroll
    for (int k = 0; k < 2; ++k) {
        int e   = sel_e[2 * n + k];
        int pos = atomicAdd(&cursors[e], 1);
        int idx = offsets[e] + pos;
        perm_tok[idx] = n;
        perm_w[idx]   = sel_w[2 * n + k];
    }
}

// ---------------------------------------------------- grouped expert GEMM ---
struct StageA { float4 v[4]; };        // 16 fp32 of x tile per thread
struct StageB { float r0[8], r1[8]; }; // 8 (even k, odd k) pairs per thread

__global__ __launch_bounds__(256)
void moe_gemm_kernel(const float* __restrict__ x, const float* __restrict__ W,
                     const float* __restrict__ bias,
                     const int* __restrict__ perm_tok, const float* __restrict__ perm_w,
                     const int* __restrict__ counts, const int* __restrict__ offsets,
                     float* __restrict__ out)
{
    const int e   = blockIdx.z;
    const int cnt = counts[e];
    const int r0  = blockIdx.y * BM;
    if (r0 >= cnt) return;
    const int n0   = blockIdx.x * BN;
    const int base = offsets[e];

    __shared__ unsigned short As[2][BM * LDSS];  // x tile [m][k], bf16
    __shared__ unsigned short Bs[2][BN * LDSS];  // W tile [n][k], bf16 (transposed)
    __shared__ int   s_tok[BM];
    __shared__ float s_w[BM];

    const int t = threadIdx.x;
    if (t < BM) {
        int r = r0 + t;
        if (r < cnt) { s_tok[t] = perm_tok[base + r]; s_w[t] = perm_w[base + r]; }
        else         { s_tok[t] = -1;                 s_w[t] = 0.f; }
    }
    __syncthreads();

    const int lane = t & 31, wave = t >> 5;
    const int wm = wave & 3;           // 4 row groups of 32
    const int wn = wave >> 2;          // 2 col groups of 64
    const int l15 = lane & 15, lhi = lane >> 4;

    // staging coordinates: one n per lane (coalesced loads, conflict-light stores)
    const int bn = t & 127;
    const int bkh = (t >> 7) * 8;      // 8 k-pairs per thread
    const float* Wp = W + ((size_t)e << 20) + (size_t)n0 + bn;

    v8f zero = {0.f, 0.f, 0.f, 0.f, 0.f, 0.f, 0.f, 0.f};
    v8f acc[2][4];
    #pragma unroll
    for (int mi = 0; mi < 2; ++mi)
        #pragma unroll
        for (int ni = 0; ni < 4; ++ni) acc[mi][ni] = zero;

    StageA ra; StageB rb;

    auto loadA = [&](int k0) {
        #pragma unroll
        for (int j = 0; j < 4; ++j) {
            int li = j * 256 + t;
            int m  = li >> 3;
            int kq = (li & 7) << 2;
            int tok = s_tok[m];
            ra.v[j] = (tok >= 0) ? *(const float4*)(x + (size_t)tok * DDIM + k0 + kq)
                                 : make_float4(0.f, 0.f, 0.f, 0.f);
        }
    };
    auto storeA = [&](int buf) {
        #pragma unroll
        for (int j = 0; j < 4; ++j) {
            int li = j * 256 + t;
            int m  = li >> 3;
            int kq = (li & 7) << 2;
            *(uint2*)(&As[buf][m * LDSS + kq]) =
                make_uint2(pack2bf(ra.v[j].x, ra.v[j].y), pack2bf(ra.v[j].z, ra.v[j].w));
        }
    };
    auto loadB = [&](int k0) {
        #pragma unroll
        for (int i = 0; i < 8; ++i) {
            int kr = (bkh + i) * 2;
            rb.r0[i] = Wp[(size_t)(k0 + kr)     * DDIM];
            rb.r1[i] = Wp[(size_t)(k0 + kr + 1) * DDIM];
        }
    };
    auto storeB = [&](int buf) {
        unsigned* B32 = (unsigned*)&Bs[buf][0];
        #pragma unroll
        for (int i = 0; i < 8; ++i)   // dword (n, k-pair): matches B frag K=2v,2v+1
            B32[bn * LDSSW + bkh + i] = pack2bf(rb.r0[i], rb.r1[i]);
    };

    // ---- software pipeline: global->regs prefetch + double-buffered LDS
    loadA(0); loadB(0);
    storeA(0); storeB(0);
    loadA(BK); loadB(BK);
    __syncthreads();

    for (int ks = 0; ks < KSTEPS; ++ks) {
        const int cur = ks & 1;
        if (ks < KSTEPS - 1) { storeA(cur ^ 1); storeB(cur ^ 1); }
        if (ks < KSTEPS - 2) { loadA((ks + 2) * BK); loadB((ks + 2) * BK); }

        v16bf a[2], bf[4];
        #pragma unroll
        for (int mi = 0; mi < 2; ++mi) {
            int m = wm * 32 + mi * 16 + l15;
            a[mi] = load_bf16x16(&As[cur][m * LDSS + lhi * 8], 16);
        }
        #pragma unroll
        for (int ni = 0; ni < 4; ++ni) {
            int n = wn * 64 + ni * 16 + l15;
            bf[ni] = load_bf16x16(&Bs[cur][n * LDSS + lhi * 8], 16);
        }
        #pragma unroll
        for (int mi = 0; mi < 2; ++mi)
            #pragma unroll
            for (int ni = 0; ni < 4; ++ni)
                acc[mi][ni] = __builtin_amdgcn_wmma_f32_16x16x32_bf16(
                    false, a[mi], false, bf[ni], (short)0, acc[mi][ni], false, false);
        __syncthreads();
    }

    // ---- epilogue: out[tok, n] += w * (acc + bias)  (exactly 2 adds/element)
    #pragma unroll
    for (int ni = 0; ni < 4; ++ni) {
        int n = n0 + wn * 64 + ni * 16 + l15;
        float bv = bias[e * DDIM + n];
        #pragma unroll
        for (int mi = 0; mi < 2; ++mi) {
            #pragma unroll
            for (int r = 0; r < 8; ++r) {
                int m = wm * 32 + mi * 16 + lhi * 8 + r;   // D: M = r + 8*lanehalf
                int tok = s_tok[m];
                if (tok >= 0)
                    atomicAdd(&out[(size_t)tok * DDIM + n], s_w[m] * (acc[mi][ni][r] + bv));
            }
        }
    }
}

// ------------------------------------------------------------------- aux ----
__global__ void aux_kernel(const float* __restrict__ imp, float* __restrict__ out_aux)
{
    if (threadIdx.x == 0 && blockIdx.x == 0) {
        const float q = 1.0f / (float)NEXP;
        float s = 0.f;
        for (int e = 0; e < NEXP; ++e) {
            float im = imp[e] * (1.0f / (float)NTOK);
            s += q * (logf(q) - logf(im));
        }
        out_aux[0] = s;
    }
}

// ---------------------------------------------------------------- launch ----
extern "C" void kernel_launch(void* const* d_in, const int* in_sizes, int n_in,
                              void* d_out, int out_size, void* d_ws, size_t ws_size,
                              hipStream_t stream)
{
    (void)in_sizes; (void)n_in; (void)out_size; (void)ws_size;
    const float* x  = (const float*)d_in[0];
    const float* W  = (const float*)d_in[1];
    const float* b  = (const float*)d_in[2];
    const float* Wg = (const float*)d_in[3];
    const float* bg = (const float*)d_in[4];
    float* out = (float*)d_out;

    int* wsi      = (int*)d_ws;
    float* imp    = (float*)wsi;                 // [16]
    int* counts   = wsi + 16;                    // [16]
    int* offsets  = wsi + 32;                    // [17]
    int* cursors  = wsi + 64;                    // [16]
    int* sel_e    = wsi + 128;                   // [2N]
    float* sel_w  = (float*)(wsi + 128 + 2 * NTOK);
    int* perm_tok = wsi + 128 + 4 * NTOK;
    float* perm_w = (float*)(wsi + 128 + 6 * NTOK);

    hipMemsetAsync(d_ws, 0, 128 * sizeof(int), stream);
    hipMemsetAsync(d_out, 0, (size_t)NTOK * DDIM * sizeof(float), stream);

    gate_topk_kernel<<<NTOK / 16, 256, 0, stream>>>(x, Wg, bg, counts, imp, sel_e, sel_w);
    scan_kernel<<<1, 32, 0, stream>>>(counts, offsets);
    scatter_kernel<<<NTOK / 256, 256, 0, stream>>>(sel_e, sel_w, offsets, cursors, perm_tok, perm_w);
    moe_gemm_kernel<<<dim3(DDIM / BN, 64, NEXP), 256, 0, stream>>>(
        x, W, b, perm_tok, perm_w, counts, offsets, out);
    aux_kernel<<<1, 1, 0, stream>>>(imp, out + (size_t)NTOK * DDIM);
}
